// GPSModel_78718160601831
// MI455X (gfx1250) — matrix operand order repaired and checked by reading, not compiled, not run on hardware
//
#include <hip/hip_runtime.h>
#include <hip/hip_bf16.h>
#include <math.h>

// ---------------- problem constants ----------------
#define NN      102400      // nodes total (G*S), multiple of 16
#define GG      512         // graphs
#define SS      200         // nodes per graph
#define EE      819200      // edges
#define CC      64          // hidden
#define HH      4           // heads
#define HD      16          // head dim
#define LL      5           // layers
#define SROW    208         // row-padded seq (13 * 16)
#define SPAD    224         // K-padded seq (7 * 32)
#define NT13    13          // ceil(200/16)
#define BN_EPS  1e-5f

typedef __bf16 bf16_t;
typedef __attribute__((ext_vector_type(16))) bf16_t v16bf;
typedef __attribute__((ext_vector_type(8)))  float  v8f;
typedef unsigned int u32;
typedef __attribute__((ext_vector_type(4))) u32 v4u;
typedef __attribute__((ext_vector_type(4))) int v4i;
typedef __attribute__((ext_vector_type(8))) int v8i;

// native f32 -> bf16 (v_cvt_pk_bf16_f32 on gfx1250)
static __device__ __forceinline__ bf16_t f2bf(float f) { return (bf16_t)f; }

// TDM: DMA `nelem` bf16 from global `src` into LDS offset 0 (1-row 2D tile).
// D# group0: count=1 | lds_addr | global_addr | type=2 ; group1: data_size=2B,
// tensor_dim0 = tile_dim0 = stride0 = nelem, tensor_dim1 = tile_dim1 = 1.
static __device__ __forceinline__ void tdm_load_lds0(const bf16_t* src, int nelem) {
    unsigned long long ga = (unsigned long long)src;
    v4u g0;
    g0[0] = 1u;                                   // count=1, user mode
    g0[1] = 0u;                                   // lds_addr = 0
    g0[2] = (u32)ga;                              // global_addr[31:0]
    g0[3] = (u32)((ga >> 32) & 0x1FFFFFFull) | (2u << 30);  // addr[56:32] | type=2
    u32 n = (u32)nelem;
    v8i g1;
    g1[0] = (int)(1u << 16);                      // data_size = 1 -> 2 bytes
    g1[1] = (int)((n & 0xFFFFu) << 16);           // tensor_dim0[15:0]
    g1[2] = (int)((n >> 16) | (1u << 16));        // tensor_dim0[31:16] | tensor_dim1=1
    g1[3] = (int)((n & 0xFFFFu) << 16);           // tile_dim0
    g1[4] = 1;                                    // tile_dim1 = 1
    g1[5] = (int)n;                               // tensor_dim0_stride[31:0]
    g1[6] = 0;
    g1[7] = 0;
    v4i z4 = {};
    v8i z8 = {};
    __builtin_amdgcn_tensor_load_to_lds(g0, g1, z4, z4, z8, 0);  // 6-arg toolchain form
    __builtin_amdgcn_s_wait_tensorcnt(0);
}

// ---------------- tiny utility kernels ----------------
__global__ void k_zero(float* p, int n) {
    int i = blockIdx.x * blockDim.x + threadIdx.x;
    if (i < n) p[i] = 0.0f;
}

__global__ void k_add(const float* __restrict__ a, const float* __restrict__ b,
                      float* __restrict__ o, int n) {
    int i = blockIdx.x * blockDim.x + threadIdx.x;
    if (i < n) o[i] = a[i] + b[i];
}

// f32 -> bf16 weight conversion + swizzle into B-fragment-major order:
// element W[k][o] -> out[(((k/32)*NT + o/16)*32 + (k%16/8)*16 + o%16)*16 + (k%8 + (k%32/16)*8)]
// so each (k-step, n-tile, lane) fragment is 16 contiguous bf16 (32B) in LDS.
__global__ void k_cvt_swz(const float* __restrict__ in, bf16_t* __restrict__ out,
                          int K, int OUT) {
    int i = blockIdx.x * blockDim.x + threadIdx.x;
    if (i >= K * OUT) return;
    int k = i / OUT, o = i - k * OUT;
    int ks = k >> 5, kr = k & 31;
    int part = (kr >> 3) & 1, q = (kr & 7) | ((kr >> 4) << 3);
    int j = o >> 4, lc = o & 15;
    int NT = OUT >> 4;
    int idx = ((((ks * NT + j) << 5) + (part << 4) + lc) << 4) + q;
    out[idx] = f2bf(in[i]);
}

// ---------------- BatchNorm (training-mode batch stats) ----------------
__global__ void k_bn_stats(const float* __restrict__ X, int n_rows, int cch,
                           float* __restrict__ ssum, float* __restrict__ ssq,
                           int rows_per_block) {
    int lane_per = (cch <= 32) ? 32 : 64;
    int c  = threadIdx.x % lane_per;
    int rg = threadIdx.x / lane_per;
    int rstep = blockDim.x / lane_per;
    int r0 = blockIdx.x * rows_per_block;
    int r1 = r0 + rows_per_block; if (r1 > n_rows) r1 = n_rows;
    if (c >= cch) return;
    float s = 0.f, q = 0.f;
    for (int r = r0 + rg; r < r1; r += rstep) {
        float v = X[(size_t)r * cch + c];
        s += v; q += v * v;
    }
    unsafeAtomicAdd(&ssum[c], s);
    unsafeAtomicAdd(&ssq[c], q);
}

// stats layout: [0..63]=sum [64..127]=sumsq [128..191]=scale [192..255]=shift
__global__ void k_bn_finalize(const float* __restrict__ g, const float* __restrict__ b,
                              float* __restrict__ stats, int cch, float inv_n) {
    int c = threadIdx.x;
    if (c >= cch) return;
    float mean = stats[c] * inv_n;
    float var  = stats[64 + c] * inv_n - mean * mean;
    float sc   = g[c] * rsqrtf(var + BN_EPS);
    stats[128 + c] = sc;
    stats[192 + c] = b[c] - mean * sc;
}

__global__ void k_bn_apply(const float* __restrict__ X, float* __restrict__ Y,
                           const float* __restrict__ stats, int n, int cch) {
    int i = blockIdx.x * blockDim.x + threadIdx.x;
    if (i >= n) return;
    int c = i % cch;
    Y[i] = X[i] * stats[128 + c] + stats[192 + c];
}

// ---------------- embedding: h = [node_emb[x] | BN(pe) @ pe_lin + b] ----------------
__global__ void k_embed(const int* __restrict__ x, const float* __restrict__ pe,
                        const float* __restrict__ nemb, const float* __restrict__ plw,
                        const float* __restrict__ plb, const float* __restrict__ stats,
                        float* __restrict__ h) {
    int i = blockIdx.x * blockDim.x + threadIdx.x;
    if (i >= NN) return;
    int t = x[i];
    float* hr = h + (size_t)i * CC;
    #pragma unroll 4
    for (int c = 0; c < 48; ++c) hr[c] = nemb[t * 48 + c];
    float pn[20];
    #pragma unroll
    for (int k = 0; k < 20; ++k)
        pn[k] = pe[(size_t)i * 20 + k] * stats[128 + k] + stats[192 + k];
    #pragma unroll
    for (int j = 0; j < 16; ++j) {
        float acc = plb[j];
        #pragma unroll
        for (int k = 0; k < 20; ++k) acc += pn[k] * plw[k * 16 + j];
        hr[48 + j] = acc;
    }
}

// ---------------- GINE message + scatter: agg[dst] += relu(h[src] + edge_emb[attr]) ----------------
__global__ void k_gine_msg(const float* __restrict__ h, const float* __restrict__ eemb,
                           const int* __restrict__ ei, const int* __restrict__ ea,
                           float* __restrict__ agg) {
    int idx = blockIdx.x * blockDim.x + threadIdx.x;   // (edge, 8-ch group)
    if (idx >= EE * 8) return;
    int e   = idx >> 3;
    int grp = (idx & 7) * 8;
    int s = ei[e];
    int d = ei[EE + e];
    int a = ea[e];
    const float* hs = h    + (size_t)s * CC + grp;
    const float* ev = eemb + (size_t)a * CC + grp;
    float*       ag = agg  + (size_t)d * CC + grp;
    #pragma unroll
    for (int j = 0; j < 8; ++j) {
        float m = hs[j] + ev[j];
        m = m > 0.f ? m : 0.f;
        unsafeAtomicAdd(&ag[j], m);
    }
}

// ---------------- templated WMMA GEMM: Y = epi((X1[+X2]) @ W + bias) ----------------
// X f32 [M,K] row-major, M % 16 == 0; Wswz = bf16 fragment-major weights, DMA'd
// into LDS by the Tensor Data Mover. Each wave computes one 16xOUT row-strip.
template <int K, int OUT>
__global__ __launch_bounds__(256)
void k_gemm_wmma(const float* __restrict__ X, const float* __restrict__ X2,
                 const bf16_t* __restrict__ Wswz, const float* __restrict__ bias,
                 const float* __restrict__ res, float* __restrict__ Y,
                 int M, int relu) {
    constexpr int NT = OUT / 16;
    constexpr int KS = K / 32;
    __shared__ __align__(32) bf16_t w_lds[K * OUT];

    if (threadIdx.x < 32) tdm_load_lds0(Wswz, K * OUT);   // wave 0 drives the TDM
    __syncthreads();

    const v16bf* wf = (const v16bf*)w_lds;

    int wave    = (blockIdx.x * blockDim.x + threadIdx.x) >> 5;
    int n_waves = (gridDim.x * blockDim.x) >> 5;
    int lane = threadIdx.x & 31;
    int lc   = lane & 15;
    int hi   = lane >> 4;
    int mtiles = M >> 4;

    for (int tm = wave; tm < mtiles; tm += n_waves) {
        v8f acc[NT];
        #pragma unroll
        for (int j = 0; j < NT; ++j) acc[j] = (v8f){};

        int row = tm * 16 + lc;

        #pragma unroll
        for (int ks = 0; ks < KS; ++ks) {
            const int k0 = ks * 32;
            // A fragment (16x32 bf16): lane&15 = row; K-halves split by lane<16
            v16bf a;
            const float* xr = X + (size_t)row * K + k0 + hi * 8;
            if (X2) {
                const float* x2 = X2 + (size_t)row * K + k0 + hi * 8;
                #pragma unroll
                for (int j = 0; j < 8; ++j) {
                    a[j]     = f2bf(xr[j]      + x2[j]);
                    a[8 + j] = f2bf(xr[16 + j] + x2[16 + j]);
                }
            } else {
                #pragma unroll
                for (int j = 0; j < 8; ++j) {
                    a[j]     = f2bf(xr[j]);
                    a[8 + j] = f2bf(xr[16 + j]);
                }
            }
            // B fragments: one contiguous 32B LDS read per N-tile; A reused
            #pragma unroll
            for (int j = 0; j < NT; ++j) {
                v16bf b = wf[(ks * NT + j) * 32 + lane];
                acc[j] = __builtin_amdgcn_wmma_f32_16x16x32_bf16(false, a, false, b,
                                                                 (short)0, acc[j],
                                                                 false, false);
            }
        }
        // epilogue: C/D layout -> row = tm*16 + r + 8*hi, col = j*16 + lc
        #pragma unroll
        for (int j = 0; j < NT; ++j) {
            int col = j * 16 + lc;
            float bv = bias ? bias[col] : 0.f;
            #pragma unroll
            for (int r = 0; r < 8; ++r) {
                int orow = tm * 16 + r + hi * 8;
                float v = acc[j][r] + bv;
                if (relu) v = v > 0.f ? v : 0.f;
                if (res) v += res[(size_t)orow * OUT + col];
                Y[(size_t)orow * OUT + col] = v;
            }
        }
    }
}

// ---------------- dense per-graph multi-head attention (WMMA + LDS) ----------------
// qkv [N,192] (q|k|v per head of 16), out o [N,64]. One block = (graph, head).
// Q,K staged into LDS pre-padded to 32 K-dims / 208 rows -> branchless fragment
// loads; V staged pre-swizzled into B-fragment-major order.
__global__ __launch_bounds__(128)
void k_attn(const float* __restrict__ qkv, float* __restrict__ o) {
    int g  = blockIdx.x >> 2;
    int hd = blockIdx.x & 3;
    __shared__ __align__(32) bf16_t q_lds[SROW * 32];     // rows padded, dims 16..31 = 0
    __shared__ __align__(32) bf16_t k_lds[SROW * 32];     // rows padded, dims 16..31 = 0
    __shared__ __align__(32) bf16_t v_swz[SPAD * HD];     // fragment-major V
    __shared__ __align__(32) bf16_t probs[4][16 * SPAD];  // per-wave softmax rows

    int tid = threadIdx.x;
    const float* base = qkv + (size_t)g * SS * 192;
    for (int i = tid; i < SROW * 32; i += 128) {
        int node = i >> 5, kk = i & 31;
        bool valid = (node < SS) & (kk < 16);
        q_lds[i] = valid ? f2bf(base[(size_t)node * 192 + hd * 16 + kk])      : f2bf(0.f);
        k_lds[i] = valid ? f2bf(base[(size_t)node * 192 + 64 + hd * 16 + kk]) : f2bf(0.f);
    }
    // V fragment-major: element (ks, lane2, j) = V[ks*32 + (lane2>=16)*8 + (j&7) + (j>>3)*16][lane2&15]
    for (int i = tid; i < SPAD * HD; i += 128) {
        int ks2 = i >> 9, lane2 = (i >> 4) & 31, j = i & 15;
        int node = ks2 * 32 + (lane2 >> 4) * 8 + (j & 7) + (j >> 3) * 16;
        int d    = lane2 & 15;
        v_swz[i] = (node < SS) ? f2bf(base[(size_t)node * 192 + 128 + hd * 16 + d])
                               : f2bf(0.f);
    }
    __syncthreads();

    const v16bf* kf = (const v16bf*)k_lds;
    const v16bf* vf = (const v16bf*)v_swz;

    int w = tid >> 5, lane = tid & 31;
    int lc = lane & 15;
    int hi = lane >> 4;
    const float scale = 0.25f;   // 1/sqrt(16)

    for (int t = w; t < NT13; t += 4) {
        // Q fragment: rows t*16..t*16+15 (LDS pre-padded -> unconditional loads)
        v16bf qa;
        {
            int row = t * 16 + lc;
            const bf16_t* qr = q_lds + row * 32 + hi * 16;
            #pragma unroll
            for (int j = 0; j < 8; ++j) { qa[j] = qr[j]; qa[8 + j] = qr[8 + j]; }
        }
        // scores: S = Q @ K^T for all 13 column tiles (branchless B loads)
        v8f c[NT13];
        #pragma unroll
        for (int ntl = 0; ntl < NT13; ++ntl) {
            v16bf kb = kf[(ntl * 16 + lc) * 2 + hi];
            v8f z = {};
            c[ntl] = __builtin_amdgcn_wmma_f32_16x16x32_bf16(false, qa, false, kb,
                                                             (short)0, z, false, false);
        }
        // scale + column mask + row softmax (rows striped across 16-lane halves)
        float mx[8], sm[8];
        #pragma unroll
        for (int r = 0; r < 8; ++r) {
            float m = -1e30f;
            #pragma unroll
            for (int ntl = 0; ntl < NT13; ++ntl) {
                int coln = ntl * 16 + lc;
                float sv = (coln < SS) ? c[ntl][r] * scale : -1e30f;
                c[ntl][r] = sv;
                m = fmaxf(m, sv);
            }
            #pragma unroll
            for (int off = 8; off > 0; off >>= 1)
                m = fmaxf(m, __shfl_xor(m, off, 16));
            mx[r] = m;
        }
        #pragma unroll
        for (int r = 0; r < 8; ++r) {
            float s = 0.f;
            #pragma unroll
            for (int ntl = 0; ntl < NT13; ++ntl) {
                float p = __expf(c[ntl][r] - mx[r]);
                c[ntl][r] = p;
                s += p;
            }
            #pragma unroll
            for (int off = 8; off > 0; off >>= 1)
                s += __shfl_xor(s, off, 16);
            sm[r] = 1.f / s;
        }
        // stash probabilities to LDS (C-frag -> row-major bf16), pad cols 208..223
        bf16_t* pw = &probs[w][0];
        #pragma unroll
        for (int ntl = 0; ntl < NT13; ++ntl) {
            #pragma unroll
            for (int r = 0; r < 8; ++r)
                pw[(r + hi * 8) * SPAD + ntl * 16 + lc] = f2bf(c[ntl][r] * sm[r]);
        }
        #pragma unroll
        for (int r = 0; r < 8; ++r)
            pw[(r + hi * 8) * SPAD + 208 + lc] = f2bf(0.f);

        // O = P @ V  (K dim padded to 224 = 7 x 32)
        v8f oc = {};
        #pragma unroll
        for (int ks = 0; ks < SPAD / 32; ++ks) {
            const int k0 = ks * 32;
            v16bf pa;
            const bf16_t* pr = pw + lc * SPAD + k0 + hi * 8;
            #pragma unroll
            for (int j = 0; j < 8; ++j) { pa[j] = pr[j]; pa[8 + j] = pr[16 + j]; }
            v16bf vb = vf[ks * 32 + lane];
            oc = __builtin_amdgcn_wmma_f32_16x16x32_bf16(false, pa, false, vb,
                                                         (short)0, oc, false, false);
        }
        #pragma unroll
        for (int r = 0; r < 8; ++r) {
            int rr = t * 16 + r + hi * 8;
            if (rr < SS)
                o[((size_t)g * SS + rr) * CC + hd * 16 + lc] = oc[r];
        }
    }
}

// ---------------- pooling (batch is contiguous) + readout head ----------------
__global__ void k_pool(const float* __restrict__ h, float* __restrict__ hg) {
    int g = blockIdx.x, c = threadIdx.x;     // block=64
    const float* base = h + (size_t)g * SS * CC + c;
    float s = 0.f;
    for (int r = 0; r < SS; ++r) s += base[(size_t)r * CC];
    hg[g * CC + c] = s;
}

__global__ void k_head(const float* __restrict__ hg,
                       const float* __restrict__ w1, const float* __restrict__ b1,
                       const float* __restrict__ w2, const float* __restrict__ b2,
                       const float* __restrict__ w3, const float* __restrict__ b3,
                       float* __restrict__ out) {
    __shared__ float y0[64], y1[32], y2[16];
    int g = blockIdx.x, t = threadIdx.x;     // block=64
    y0[t] = hg[g * CC + t];
    __syncthreads();
    if (t < 32) {
        float a = b1[t];
        #pragma unroll 8
        for (int k = 0; k < 64; ++k) a += y0[k] * w1[k * 32 + t];
        y1[t] = a > 0.f ? a : 0.f;
    }
    __syncthreads();
    if (t < 16) {
        float a = b2[t];
        #pragma unroll
        for (int k = 0; k < 32; ++k) a += y1[k] * w2[k * 16 + t];
        y2[t] = a > 0.f ? a : 0.f;
    }
    __syncthreads();
    if (t == 0) {
        float a = b3[0];
        #pragma unroll
        for (int k = 0; k < 16; ++k) a += y2[k] * w3[k];
        out[g] = a;
    }
}

// ---------------- host orchestration ----------------
static inline int cdiv(int a, int b) { return (a + b - 1) / b; }

extern "C" void kernel_launch(void* const* d_in, const int* in_sizes, int n_in,
                              void* d_out, int out_size, void* d_ws, size_t ws_size,
                              hipStream_t stream) {
    (void)in_sizes; (void)n_in; (void)out_size; (void)ws_size;

    // inputs (setup_inputs dict order), params in jax-pytree (sorted-key) order
    const int*   x   = (const int*)d_in[0];
    const float* pe  = (const float*)d_in[1];
    const int*   ei  = (const int*)d_in[2];
    const int*   ea  = (const int*)d_in[3];
    // d_in[4] = batch (unused: contiguous equal-size graphs)
    const float* edge_emb = (const float*)d_in[5];
    // layers: d_in[6 + l*20 + k], k in sorted key order:
    // 0 attn_in_b 1 attn_in_w 2 attn_out_b 3 attn_out_w 4 gine_b1 5 gine_b2
    // 6 gine_w1 7 gine_w2 8 mlp_b1 9 mlp_b2 10 mlp_w1 11 mlp_w2
    // 12 n1_b 13 n1_g 14 n2_b 15 n2_g 16 n3_b 17 n3_g 18 on_b 19 on_g
    auto LPP = [&](int l, int k) -> const float* { return (const float*)d_in[6 + l * 20 + k]; };
    const float* node_emb  = (const float*)d_in[106];
    const float* out_b1    = (const float*)d_in[107];
    const float* out_b2    = (const float*)d_in[108];
    const float* out_b3    = (const float*)d_in[109];
    const float* out_w1    = (const float*)d_in[110];
    const float* out_w2    = (const float*)d_in[111];
    const float* out_w3    = (const float*)d_in[112];
    const float* pe_lin_b  = (const float*)d_in[113];
    const float* pe_lin_w  = (const float*)d_in[114];
    const float* pe_norm_b = (const float*)d_in[115];
    const float* pe_norm_g = (const float*)d_in[116];

    // workspace layout (f32)
    float* ws = (float*)d_ws;
    size_t off = 0;
    float* h     = ws + off; off += (size_t)NN * CC;
    float* agg   = ws + off; off += (size_t)NN * CC;   // also reused as t3
    float* P0    = ws + off; off += (size_t)NN * 192;  // qkv / z1 / m1
    float* t1    = ws + off; off += (size_t)NN * CC;   // gine branch
    float* t2    = ws + off; off += (size_t)NN * CC;   // attn branch
    float* outb  = ws + off; off += (size_t)NN * CC;   // combined
    float* obuf  = ws + off; off += (size_t)NN * CC;   // attention raw output
    float* hg    = ws + off; off += (size_t)GG * CC;
    float* stats = ws + off; off += 256;
    bf16_t* wscr = (bf16_t*)(ws + off); off += (128 * 192 * 2 + 4) / 4;  // bf16 weight scratch

    auto gemm = [&](const float* X, const float* X2, const float* W,
                    const float* bias, const float* res, float* Y,
                    int M, int K, int OUT, int relu) {
        k_cvt_swz<<<cdiv(K * OUT, 256), 256, 0, stream>>>(W, wscr, K, OUT);
        int blocks = cdiv(cdiv(M, 16), 8);          // 8 waves / 256-thread block
        if (K == 64 && OUT == 64)
            k_gemm_wmma<64, 64><<<blocks, 256, 0, stream>>>(X, X2, wscr, bias, res, Y, M, relu);
        else if (K == 64 && OUT == 128)
            k_gemm_wmma<64, 128><<<blocks, 256, 0, stream>>>(X, X2, wscr, bias, res, Y, M, relu);
        else if (K == 64 && OUT == 192)
            k_gemm_wmma<64, 192><<<blocks, 256, 0, stream>>>(X, X2, wscr, bias, res, Y, M, relu);
        else
            k_gemm_wmma<128, 64><<<blocks, 256, 0, stream>>>(X, X2, wscr, bias, res, Y, M, relu);
    };
    auto bn = [&](const float* X, float* Y, const float* g, const float* b,
                  int rows, int cch) {
        k_zero<<<1, 128, 0, stream>>>(stats, 128);
        k_bn_stats<<<cdiv(rows, 512), 256, 0, stream>>>(X, rows, cch, stats, stats + 64, 512);
        k_bn_finalize<<<1, 64, 0, stream>>>(g, b, stats, cch, 1.0f / (float)rows);
        k_bn_apply<<<cdiv(rows * cch, 256), 256, 0, stream>>>(X, Y, stats, rows * cch, cch);
    };

    // --- embedding: BN(pe) -> linear, concat node_emb ---
    k_zero<<<1, 128, 0, stream>>>(stats, 128);
    k_bn_stats<<<cdiv(NN, 512), 256, 0, stream>>>(pe, NN, 20, stats, stats + 64, 512);
    k_bn_finalize<<<1, 64, 0, stream>>>(pe_norm_g, pe_norm_b, stats, 20, 1.0f / (float)NN);
    k_embed<<<cdiv(NN, 256), 256, 0, stream>>>(x, pe, node_emb, pe_lin_w, pe_lin_b, stats, h);

    for (int l = 0; l < LL; ++l) {
        // --- GINE branch ---
        k_zero<<<cdiv(NN * CC, 256), 256, 0, stream>>>(agg, NN * CC);
        k_gine_msg<<<cdiv(EE * 8, 256), 256, 0, stream>>>(h, edge_emb, ei, ea, agg);
        gemm(h, agg, LPP(l, 6), LPP(l, 4), nullptr, P0, NN, CC, CC, 1);      // relu((h+agg)W1+b1)
        gemm(P0, nullptr, LPP(l, 7), LPP(l, 5), h, t1, NN, CC, CC, 0);       // zW2+b2 + h
        bn(t1, t1, LPP(l, 13), LPP(l, 12), NN, CC);                          // n1 -> h_local

        // --- attention branch ---
        gemm(h, nullptr, LPP(l, 1), LPP(l, 0), nullptr, P0, NN, CC, 3 * CC, 0);  // qkv
        k_attn<<<GG * HH, 128, 0, stream>>>(P0, obuf);
        gemm(obuf, nullptr, LPP(l, 3), LPP(l, 2), h, t2, NN, CC, CC, 0);     // o W + b + h
        bn(t2, t2, LPP(l, 15), LPP(l, 14), NN, CC);                          // n2 -> h_attn

        // --- combine + FFN ---
        k_add<<<cdiv(NN * CC, 256), 256, 0, stream>>>(t1, t2, outb, NN * CC);
        gemm(outb, nullptr, LPP(l, 10), LPP(l, 8), nullptr, P0, NN, CC, 2 * CC, 1); // relu
        gemm(P0, nullptr, LPP(l, 11), LPP(l, 9), outb, agg, NN, 2 * CC, CC, 0);    // + residual
        bn(agg, agg, LPP(l, 17), LPP(l, 16), NN, CC);                        // n3
        bn(agg, h, LPP(l, 19), LPP(l, 18), NN, CC);                          // outer norm -> h
    }

    // --- pooling + readout head ---
    k_pool<<<GG, 64, 0, stream>>>(h, hg);
    k_head<<<GG, 64, 0, stream>>>(hg, out_w1, out_b1, out_w2, out_b2, out_w3, out_b3,
                                  (float*)d_out);
}
